// CrossLayerTranscoder_34351148434220
// MI455X (gfx1250) — matrix operand order repaired and checked by reading, not compiled, not run on hardware
//
#include <hip/hip_runtime.h>
#include <hip/hip_bf16.h>

// CrossLayerTranscoder for MI455X (gfx1250, wave32, WMMA, async-LDS).
// Stage 1: feats[l] = relu(x[l] @ W_enc[l]^T + b_enc[l])  -> bf16 in d_ws (128 MB)
// Stage 2: recon[j] = sum_{l<=j} feats[l] @ W_dec[l,:,j,:] + b_dec[j] -> fp32 d_out
// WG tile 128x128, 8 waves (4x2), wave tile 32x64 -> 8 v_wmma per K=32 step.

#define L_ 8
#define B_ 2048
#define D_ 768
#define F_ 4096

typedef __attribute__((ext_vector_type(16))) __bf16 v16bf;
typedef __attribute__((ext_vector_type(2)))  __bf16 v2bf;
typedef __attribute__((ext_vector_type(8)))  float  v8f;
typedef __attribute__((ext_vector_type(4)))  int    v4i;

#if defined(__has_builtin)
#if __has_builtin(__builtin_amdgcn_global_load_async_to_lds_b128)
#define HAVE_ASYNC_LDS 1
#endif
#if __has_builtin(__builtin_amdgcn_cvt_pk_bf16_f32)
#define HAVE_CVT_PK_BF16 1
#endif
#endif

union Frag {
    unsigned int u[8];
    v16bf        v;
};

__device__ __forceinline__ unsigned int pack_bf16x2(float a, float b) {
#if defined(HAVE_CVT_PK_BF16)
    union { v2bf v; unsigned int u; } r;
    r.v = __builtin_amdgcn_cvt_pk_bf16_f32(a, b);
    return r.u;
#else
    unsigned int ua = __float_as_uint(a);
    unsigned int ub = __float_as_uint(b);
    ua = (ua + 0x7FFFu + ((ua >> 16) & 1u)) >> 16;   // round-to-nearest-even
    ub = (ub + 0x7FFFu + ((ub >> 16) & 1u)) >> 16;
    return (ua & 0xFFFFu) | (ub << 16);
#endif
}

__device__ __forceinline__ unsigned short f32_to_bf16(float a) {
#if defined(HAVE_CVT_PK_BF16)
    union { v2bf v; unsigned short s[2]; } r;
    r.v = __builtin_amdgcn_cvt_pk_bf16_f32(a, a);
    return r.s[0];
#else
    unsigned int ua = __float_as_uint(a);
    ua = (ua + 0x7FFFu + ((ua >> 16) & 1u)) >> 16;
    return (unsigned short)ua;
#endif
}

__device__ __forceinline__ void wait_async0() {
#if defined(HAVE_ASYNC_LDS)
#if __has_builtin(__builtin_amdgcn_s_wait_asynccnt)
    __builtin_amdgcn_s_wait_asynccnt(0);
#else
    asm volatile("s_wait_asynccnt 0x0" ::: "memory");
#endif
#endif
}

// Rows padded to 20 dwords (80B): keeps 16B alignment for b128 async writes /
// ds_load_b128 fragment reads and is bank-conflict-free for the gathers
// (20*m mod 64 distinct for m=0..15).
#define APAD 20

// ---------------------------------------------------------------------------
// Encode: per layer l, C[128x128] tile of feats = relu(x @ W_enc^T + b_enc)
// ---------------------------------------------------------------------------
__global__ __launch_bounds__(256)
void encode_kernel(const float* __restrict__ x,
                   const float* __restrict__ W_enc,
                   const float* __restrict__ b_enc,
                   unsigned short* __restrict__ feats) {
    __shared__ __align__(16) unsigned int Asl[128][APAD];  // 128 b-rows x 32 k (bf16 pairs)
    __shared__ __align__(16) unsigned int Bsl[128][APAD];  // 128 f-rows x 32 k (bf16 pairs)

    const int tid  = threadIdx.x;
    const int wid  = tid >> 5;
    const int lane = tid & 31;
    const int half = lane >> 4;
    const int l16  = lane & 15;
    const int l    = blockIdx.z;
    const int m0   = blockIdx.x * 128;    // token tile
    const int n0   = blockIdx.y * 128;    // feature tile
    const int wm   = (wid >> 1) * 32;     // wave grid 4x2, wave tile 32x64
    const int wn   = (wid & 1) * 64;

    v8f acc[2][4];
#pragma unroll
    for (int i = 0; i < 2; ++i)
#pragma unroll
        for (int jt = 0; jt < 4; ++jt)
#pragma unroll
            for (int r = 0; r < 8; ++r) acc[i][jt][r] = 0.0f;

    for (int k0 = 0; k0 < D_; k0 += 32) {
        // A tile: x[l, m0..+127, k0..+31] fp32 -> packed bf16 (2048 dwords)
#pragma unroll
        for (int t = 0; t < 8; ++t) {
            int idx = tid + t * 256;
            int kk = idx & 15, m = idx >> 4;
            const float2 f2 = *(const float2*)(x + ((size_t)l * B_ + (m0 + m)) * D_ + (k0 + 2 * kk));
            Asl[m][kk] = pack_bf16x2(f2.x, f2.y);
        }
        // B tile: W_enc[l, n0..+127, k0..+31] (k contiguous per f-row)
#pragma unroll
        for (int t = 0; t < 8; ++t) {
            int idx = tid + t * 256;
            int kk = idx & 15, n = idx >> 4;
            const float2 f2 = *(const float2*)(W_enc + ((size_t)l * F_ + (n0 + n)) * D_ + (k0 + 2 * kk));
            Bsl[n][kk] = pack_bf16x2(f2.x, f2.y);
        }
        __syncthreads();

        Frag a[2], b[4];
        // A frag (16-bit 16x32): lanes 0-15 K=0..15; lanes 16-31 K=16..31
#pragma unroll
        for (int i = 0; i < 2; ++i)
#pragma unroll
            for (int v = 0; v < 8; ++v)
                a[i].u[v] = Asl[wm + i * 16 + l16][v + half * 8];
        // B frag: lanes 0-15 K=0..15; lanes 16-31 K=16..31
#pragma unroll
        for (int jt = 0; jt < 4; ++jt)
#pragma unroll
            for (int v = 0; v < 8; ++v)
                b[jt].u[v] = Bsl[wn + jt * 16 + l16][v + half * 8];

#pragma unroll
        for (int i = 0; i < 2; ++i)
#pragma unroll
            for (int jt = 0; jt < 4; ++jt)
                acc[i][jt] = __builtin_amdgcn_wmma_f32_16x16x32_bf16(
                    false, a[i].v, false, b[jt].v, (short)0, acc[i][jt], false, false);
        __syncthreads();
    }

    // Epilogue: bias + ReLU -> bf16 feats.
    // C/D layout: vgpr r: lanes 0-15 -> M=r, N=lane; lanes 16-31 -> M=8+r, N=lane-16
#pragma unroll
    for (int i = 0; i < 2; ++i)
#pragma unroll
        for (int jt = 0; jt < 4; ++jt)
#pragma unroll
            for (int r = 0; r < 8; ++r) {
                int brow = m0 + wm + i * 16 + half * 8 + r;
                int fcol = n0 + wn + jt * 16 + l16;
                float v = acc[i][jt][r] + b_enc[(size_t)l * F_ + fcol];
                v = v > 0.0f ? v : 0.0f;
                feats[((size_t)l * B_ + brow) * F_ + fcol] = f32_to_bf16(v);
            }
}

// ---------------------------------------------------------------------------
// Decode: recon[j] = sum_{l<=j} feats[l] @ W_dec[l,:,j,:] + b_dec[j]
// grid: (B/128, D/128, L); block 256.
// ---------------------------------------------------------------------------
__global__ __launch_bounds__(256)
void decode_kernel(const unsigned short* __restrict__ feats,
                   const float* __restrict__ W_dec,
                   const float* __restrict__ b_dec,
                   float* __restrict__ out) {
    __shared__ __align__(16) unsigned int Asl[128][APAD];  // 128 tokens x 32 f (bf16 pairs)
    __shared__ __align__(16) unsigned int Bsl[128][APAD];  // 128 d     x 32 f (bf16 pairs)

    const int tid  = threadIdx.x;
    const int wid  = tid >> 5;
    const int lane = tid & 31;
    const int half = lane >> 4;
    const int l16  = lane & 15;
    const int jz   = blockIdx.z;          // target layer
    const int m0   = blockIdx.x * 128;    // token tile
    const int n0   = blockIdx.y * 128;    // d_model tile
    const int wm   = (wid >> 1) * 32;
    const int wn   = (wid & 1) * 64;

    const unsigned int* featsU = (const unsigned int*)feats;
    (void)featsU;

    v8f acc[2][4];
#pragma unroll
    for (int i = 0; i < 2; ++i)
#pragma unroll
        for (int jt = 0; jt < 4; ++jt)
#pragma unroll
            for (int r = 0; r < 8; ++r) acc[i][jt][r] = 0.0f;

    for (int l = 0; l <= jz; ++l) {
        for (int k0 = 0; k0 < F_; k0 += 32) {
            // ---- A tile: feats already bf16 -> raw global->LDS copy ----
#if defined(HAVE_ASYNC_LDS)
            // 128 rows x 64B = 512 x b128 chunks; 2 per thread, async engine copy.
#pragma unroll
            for (int t = 0; t < 2; ++t) {
                int idx = tid + t * 256;
                int m = idx >> 2, c = idx & 3;            // 4 x 16B chunks per row
                unsigned short* g = const_cast<unsigned short*>(
                    feats + ((size_t)l * B_ + (m0 + m)) * F_ + k0 + c * 8);
                __attribute__((address_space(1))) v4i* gsrc =
                    (__attribute__((address_space(1))) v4i*)g;
                __attribute__((address_space(3))) v4i* ldst =
                    (__attribute__((address_space(3))) v4i*)&Asl[m][c * 4];
                __builtin_amdgcn_global_load_async_to_lds_b128(gsrc, ldst, 0, 0);
            }
#else
#pragma unroll
            for (int t = 0; t < 8; ++t) {
                int idx = tid + t * 256;
                int kk = idx & 15, m = idx >> 4;
                Asl[m][kk] = featsU[(((size_t)l * B_ + (m0 + m)) * F_ + k0) / 2 + kk];
            }
#endif
            // ---- B tile: W_dec[l, f, jz, n0+n]; coalesced over d, strided over f ----
#pragma unroll
            for (int t = 0; t < 8; ++t) {
                int idx = tid + t * 256;
                int n = idx & 127, kk = idx >> 7;         // n: d offset, kk: f pair
                int f = k0 + 2 * kk;
                size_t base = (((size_t)l * F_ + f) * L_ + jz) * (size_t)D_ + (n0 + n);
                float lo = W_dec[base];
                float hi = W_dec[base + (size_t)L_ * D_];
                Bsl[n][kk] = pack_bf16x2(lo, hi);
                if (t == 0 && k0 + 32 < F_)               // prefetch next K slab
                    __builtin_prefetch(W_dec + base + (size_t)32 * L_ * D_, 0, 0);
            }
            wait_async0();
            __syncthreads();

            Frag a[2], b[4];
#pragma unroll
            for (int i = 0; i < 2; ++i)
#pragma unroll
                for (int v = 0; v < 8; ++v)
                    a[i].u[v] = Asl[wm + i * 16 + l16][v + half * 8];
#pragma unroll
            for (int jt = 0; jt < 4; ++jt)
#pragma unroll
                for (int v = 0; v < 8; ++v)
                    b[jt].u[v] = Bsl[wn + jt * 16 + l16][v + half * 8];

#pragma unroll
            for (int i = 0; i < 2; ++i)
#pragma unroll
                for (int jt = 0; jt < 4; ++jt)
                    acc[i][jt] = __builtin_amdgcn_wmma_f32_16x16x32_bf16(
                        false, a[i].v, false, b[jt].v, (short)0, acc[i][jt], false, false);
            __syncthreads();
        }
    }

    // Epilogue: + b_dec, fp32 store
#pragma unroll
    for (int i = 0; i < 2; ++i)
#pragma unroll
        for (int jt = 0; jt < 4; ++jt)
#pragma unroll
            for (int r = 0; r < 8; ++r) {
                int brow = m0 + wm + i * 16 + half * 8 + r;
                int dcol = n0 + wn + jt * 16 + l16;
                out[((size_t)jz * B_ + brow) * D_ + dcol] =
                    acc[i][jt][r] + b_dec[(size_t)jz * D_ + dcol];
            }
}

extern "C" void kernel_launch(void* const* d_in, const int* in_sizes, int n_in,
                              void* d_out, int out_size, void* d_ws, size_t ws_size,
                              hipStream_t stream) {
    const float* x     = (const float*)d_in[0];
    const float* W_enc = (const float*)d_in[1];
    const float* b_enc = (const float*)d_in[2];
    const float* W_dec = (const float*)d_in[3];
    const float* b_dec = (const float*)d_in[4];
    float* out = (float*)d_out;

    // feats (bf16) scratch: L*B*F*2 = 128 MiB in d_ws
    unsigned short* feats = (unsigned short*)d_ws;

    dim3 blk(256);
    dim3 ge(B_ / 128, F_ / 128, L_);  // 16 x 32 x 8
    encode_kernel<<<ge, blk, 0, stream>>>(x, W_enc, b_enc, feats);

    dim3 gd(B_ / 128, D_ / 128, L_); // 16 x 6 x 8
    decode_kernel<<<gd, blk, 0, stream>>>(feats, W_dec, b_dec, out);
}
